// Transformer_82119774699897
// MI455X (gfx1250) — compile-verified, matrix operand order checked by reading, not run on hardware
//
#include <hip/hip_runtime.h>
#include <hip/hip_bf16.h>

// ---------------------------------------------------------------------------
// Sinkhorn Transformer forward for MI455X (gfx1250, wave32, WMMA).
// Heavy GEMMs run through v_wmma_f32_16x16x32_bf16; GEMM tiles are staged
// into LDS with GLOBAL_LOAD_ASYNC_TO_LDS_B128 (ASYNCcnt) when available.
// ---------------------------------------------------------------------------

typedef __bf16 bf16;
typedef __bf16 v16bf __attribute__((ext_vector_type(16)));
typedef float  v8f   __attribute__((ext_vector_type(8)));
typedef int    v4i   __attribute__((ext_vector_type(4)));

#if defined(__AMDGCN__) && \
    __has_builtin(__builtin_amdgcn_global_load_async_to_lds_b128) && \
    __has_builtin(__builtin_amdgcn_s_wait_asynccnt)
#define USE_ASYNC_COPY 1
// Builtin signature (from clang diagnostics): (v4i AS1*, v4i AS3*, imm, imm)
typedef v4i __attribute__((address_space(1)))* gv4i_p;
typedef v4i __attribute__((address_space(3)))* lv4i_p;
#else
#define USE_ASYNC_COPY 0
#endif

#define DBG_B   2
#define DBG_N   2160
#define DBG_M   (DBG_B * DBG_N)   // 4320 tokens
#define DBG_D   512
#define DBG_H   8
#define DBG_DH  64
#define DBG_BS  120
#define DBG_NB  18
#define DBG_L   6

// ---------------------------------------------------------------------------
// WMMA fragment loaders (16-bit operands, wave32).
// A operand from row-major [16][K] tile: lane L<16 holds row M=L with
// K = {0..7,16..23}; lane L>=16 holds row M=L-16 with K = {8..15,24..31}.
// v16bf element 2j/2j+1 = low/high half of VGPR j.
// ---------------------------------------------------------------------------
__device__ inline v16bf frag_rowmajor(const bf16* __restrict__ base, int ld) {
  const int lane = threadIdx.x & 31;
  const int row  = lane & 15;
  const int kb   = (lane >> 4) << 3;
  const bf16* p  = base + (size_t)row * ld + kb;
  v16bf f;
#pragma unroll
  for (int j = 0; j < 8; ++j) {
    const int k = (j < 4) ? (2 * j) : (16 + 2 * (j - 4));
    f[2 * j]     = p[k];
    f[2 * j + 1] = p[k + 1];
  }
  return f;
}

// B operand from K-major [32][ld] tile (element (k,n) at base[k*ld + n]);
// lane holds column n = lane&15 relative to base.
__device__ inline v16bf frag_kxn(const bf16* __restrict__ base, int ld) {
  const int lane = threadIdx.x & 31;
  const int kb   = (lane >> 4) << 3;
  const bf16* p  = base + (lane & 15);
  v16bf f;
#pragma unroll
  for (int j = 0; j < 8; ++j) {
    const int k = kb + ((j < 4) ? (2 * j) : (16 + 2 * (j - 4)));
    f[2 * j]     = p[(size_t)k * ld];
    f[2 * j + 1] = p[(size_t)(k + 1) * ld];
  }
  return f;
}

// ---------------------------------------------------------------------------
// f32 -> bf16 conversion (weights, once per launch)
// ---------------------------------------------------------------------------
__global__ void cvt_kernel(const float* __restrict__ src, bf16* __restrict__ dst, int n) {
  int i = blockIdx.x * blockDim.x + threadIdx.x;
  if (i < n) dst[i] = (bf16)src[i];
}

// ---------------------------------------------------------------------------
// Patch embedding: conv3d kernel=stride=8 over x[2,4,96,96,120] -> h[B,N,512]
// plus bias and positional embedding. One block per patch; patch cached in LDS.
// ---------------------------------------------------------------------------
__global__ __launch_bounds__(256) void patch_kernel(
    const float* __restrict__ x, const float* __restrict__ Wp,
    const float* __restrict__ bp, const float* __restrict__ pos,
    float* __restrict__ h) {
  __shared__ float patch[2048];
  const int blk = blockIdx.x;            // b*2160 + n
  const int n = blk % DBG_N, b = blk / DBG_N;
  const int pd = n / 180, rem = n % 180, ph = rem / 15, pw = rem % 15;
  for (int p = threadIdx.x; p < 2048; p += 256) {
    const int kw = p & 7, t1 = p >> 3;
    const int kh = t1 & 7, t2 = t1 >> 3;
    const int kd = t2 & 7, c = t2 >> 3;
    const size_t xi = (((size_t)(b * 4 + c) * 96 + (8 * pd + kd)) * 96 +
                       (8 * ph + kh)) * 120 + (8 * pw + kw);
    patch[p] = x[xi];
  }
  __syncthreads();
  for (int o = threadIdx.x; o < DBG_D; o += 256) {
    const float* w = Wp + (size_t)o * 2048;
    float acc = 0.f;
#pragma unroll 8
    for (int p = 0; p < 2048; ++p) acc += patch[p] * w[p];
    h[(size_t)blk * DBG_D + o] = acc + bp[o] + pos[(size_t)n * DBG_D + o];
  }
}

// ---------------------------------------------------------------------------
// LayerNorm over D=512, emits bf16. One block (8 waves) per token.
// ---------------------------------------------------------------------------
__global__ __launch_bounds__(256) void ln_kernel(
    const float* __restrict__ h, const float* __restrict__ g,
    const float* __restrict__ bta, bf16* __restrict__ out) {
  const int row = blockIdx.x;
  const float* xr = h + (size_t)row * DBG_D;
  const int tid = threadIdx.x;
  const float a = xr[tid], c = xr[tid + 256];
  float s = a + c, s2 = a * a + c * c;
#pragma unroll
  for (int m = 1; m < 32; m <<= 1) {
    s  += __shfl_xor(s, m, 32);
    s2 += __shfl_xor(s2, m, 32);
  }
  __shared__ float ls[8], ls2[8];
  if ((tid & 31) == 0) { ls[tid >> 5] = s; ls2[tid >> 5] = s2; }
  __syncthreads();
  s = 0.f; s2 = 0.f;
#pragma unroll
  for (int w = 0; w < 8; ++w) { s += ls[w]; s2 += ls2[w]; }
  const float mean = s * (1.f / DBG_D);
  const float var  = s2 * (1.f / DBG_D) - mean * mean;
  const float r    = rsqrtf(var + 1e-5f);
  out[(size_t)row * DBG_D + tid]       = (bf16)((a - mean) * r * g[tid] + bta[tid]);
  out[(size_t)row * DBG_D + tid + 256] = (bf16)((c - mean) * r * g[tid + 256] + bta[tid + 256]);
}

// ---------------------------------------------------------------------------
// Generic bf16 GEMM with f32 WMMA accumulation.
// C[M,N] = A[M,K] @ B[K,N]; block tile 64x64, 4 waves, each wave 2x2 WMMA tiles.
// Tiles staged to LDS with async global->LDS copies (ASYNCcnt) when supported.
// EPI: 0 = store bf16, 1 = +bias, tanh-GELU, store bf16, 2 = +bias, += f32 out.
// ---------------------------------------------------------------------------
template <int EPI>
__global__ __launch_bounds__(128) void gemm_bf16(
    const bf16* __restrict__ A, const bf16* __restrict__ Bw,
    const float* __restrict__ bias, bf16* __restrict__ Obf,
    float* __restrict__ Ores, int M, int N, int K) {
  __shared__ bf16 As[64][32];     // [m][k]
  __shared__ bf16 Bs[32][64];     // [k][n]  (straight copy of B tile)
  const int tid  = threadIdx.x;
  const int wave = tid >> 5;
  const int lane = tid & 31;
  const int m0 = blockIdx.y * 64;
  const int n0 = blockIdx.x * 64;
  const int wm = (wave & 1) * 32;
  const int wn = (wave >> 1) * 32;
  v8f acc[2][2] = {};
  for (int k0 = 0; k0 < K; k0 += 32) {
#if USE_ASYNC_COPY
    // ---- async staging: memory -> LDS without touching VGPRs ----
#pragma unroll
    for (int t = 0; t < 2; ++t) {           // A tile: 64x32 bf16 = 256 x b128
      const int chunk = tid + t * 128;      // 0..255
      const int r  = chunk >> 2;
      const int c8 = (chunk & 3) << 3;
      int gm = m0 + r;
      if (gm >= M) gm = m0;                 // clamp: rows >= M never stored
      __builtin_amdgcn_global_load_async_to_lds_b128(
          (gv4i_p)(A + (size_t)gm * K + k0 + c8),
          (lv4i_p)&As[r][c8], 0, 0);
    }
#pragma unroll
    for (int t = 0; t < 2; ++t) {           // B tile: 32x64 bf16 = 256 x b128
      const int chunk = tid + t * 128;
      const int kr = chunk >> 3;
      const int n8 = (chunk & 7) << 3;
      __builtin_amdgcn_global_load_async_to_lds_b128(
          (gv4i_p)(Bw + (size_t)(k0 + kr) * N + n0 + n8),
          (lv4i_p)&Bs[kr][n8], 0, 0);
    }
    __builtin_amdgcn_s_wait_asynccnt(0);
    __syncthreads();
#else
    // ---- synchronous fallback (both sides coalesced) ----
    for (int idx = tid; idx < 64 * 32; idx += 128) {
      const int r = idx >> 5, c = idx & 31;
      const int gm = m0 + r;
      As[r][c] = (gm < M) ? A[(size_t)gm * K + k0 + c] : (bf16)0.f;
    }
    for (int idx = tid; idx < 32 * 64; idx += 128) {
      const int nn = idx & 63, kr = idx >> 6;
      Bs[kr][nn] = Bw[(size_t)(k0 + kr) * N + n0 + nn];
    }
    __syncthreads();
#endif
    v16bf af[2], bfg[2];
    af[0]  = frag_rowmajor(&As[wm][0], 32);
    af[1]  = frag_rowmajor(&As[wm + 16][0], 32);
    bfg[0] = frag_kxn(&Bs[0][wn], 64);
    bfg[1] = frag_kxn(&Bs[0][wn + 16], 64);
#pragma unroll
    for (int i = 0; i < 2; ++i)
#pragma unroll
      for (int j = 0; j < 2; ++j)
        acc[i][j] = __builtin_amdgcn_wmma_f32_16x16x32_bf16(
            false, af[i], false, bfg[j], (short)0, acc[i][j], false, false);
    __syncthreads();
  }
  const int colL  = lane & 15;
  const int rhalf = (lane >> 4) * 8;
#pragma unroll
  for (int i = 0; i < 2; ++i)
#pragma unroll
    for (int j = 0; j < 2; ++j)
#pragma unroll
      for (int e = 0; e < 8; ++e) {
        const int gm = m0 + wm + i * 16 + rhalf + e;
        const int gn = n0 + wn + j * 16 + colL;
        if (gm < M) {
          float v = acc[i][j][e];
          if constexpr (EPI == 0) {
            Obf[(size_t)gm * N + gn] = (bf16)v;
          } else if constexpr (EPI == 1) {
            v += bias[gn];
            const float t = 0.7978845608028654f * (v + 0.044715f * v * v * v);
            v = 0.5f * v * (1.f + tanhf(t));
            Obf[(size_t)gm * N + gn] = (bf16)v;
          } else {
            v += bias[gn];
            Ores[(size_t)gm * N + gn] += v;
          }
        }
      }
}

// ---------------------------------------------------------------------------
// Bucket summaries: sq/sk[b,h,i,d] = mean over s of q/k. 288 blocks x 64 thr.
// ---------------------------------------------------------------------------
__global__ __launch_bounds__(64) void summ_kernel(
    const bf16* __restrict__ qkv, float* __restrict__ sq, float* __restrict__ sk) {
  const int blk = blockIdx.x;                 // (b*8+h)*18+i
  const int i = blk % DBG_NB, bh = blk / DBG_NB;
  const int hh = bh % DBG_H, b = bh / DBG_H;
  const int d = threadIdx.x;
  float aq = 0.f, ak = 0.f;
  const size_t base = ((size_t)(b * DBG_N + i * DBG_BS)) * (3 * DBG_D) + hh * DBG_DH + d;
  for (int s = 0; s < DBG_BS; ++s) {
    aq += (float)qkv[base + (size_t)s * (3 * DBG_D)];
    ak += (float)qkv[base + (size_t)s * (3 * DBG_D) + DBG_D];
  }
  sq[blk * DBG_DH + d] = aq * (1.f / DBG_BS);
  sk[blk * DBG_DH + d] = ak * (1.f / DBG_BS);
}

// ---------------------------------------------------------------------------
// Sinkhorn routing matrix: R[b,h,18,18]. One wave per (b,h); 7 iterations
// of log-space row/col normalization, then exp. 16 blocks x 32 threads.
// ---------------------------------------------------------------------------
__global__ __launch_bounds__(32) void sinkhorn_kernel(
    const float* __restrict__ sq, const float* __restrict__ sk, float* __restrict__ R) {
  __shared__ float r[DBG_NB * DBG_NB];
  const int bh = blockIdx.x;
  const int lane = threadIdx.x;
  for (int t = lane; t < DBG_NB * DBG_NB; t += 32) {
    const int i = t / DBG_NB, j = t % DBG_NB;
    const float* a = sq + (size_t)(bh * DBG_NB + i) * DBG_DH;
    const float* c = sk + (size_t)(bh * DBG_NB + j) * DBG_DH;
    float dot = 0.f;
#pragma unroll 8
    for (int d = 0; d < DBG_DH; ++d) dot += a[d] * c[d];
    dot *= 0.125f;                                        // dh^-0.5
    r[t] = logf(fmaxf(dot, 0.f) + 1e-6f) * (1.f / 0.75f); // log(relu+eps)/TEMP
  }
  __syncthreads();
  for (int it = 0; it < 7; ++it) {
    if (lane < DBG_NB) {
      float m = -1e30f;
      for (int j = 0; j < DBG_NB; ++j) m = fmaxf(m, r[lane * DBG_NB + j]);
      float s = 0.f;
      for (int j = 0; j < DBG_NB; ++j) s += __expf(r[lane * DBG_NB + j] - m);
      const float l = m + __logf(s);
      for (int j = 0; j < DBG_NB; ++j) r[lane * DBG_NB + j] -= l;
    }
    __syncthreads();
    if (lane < DBG_NB) {
      float m = -1e30f;
      for (int ii = 0; ii < DBG_NB; ++ii) m = fmaxf(m, r[ii * DBG_NB + lane]);
      float s = 0.f;
      for (int ii = 0; ii < DBG_NB; ++ii) s += __expf(r[ii * DBG_NB + lane] - m);
      const float l = m + __logf(s);
      for (int ii = 0; ii < DBG_NB; ++ii) r[ii * DBG_NB + lane] -= l;
    }
    __syncthreads();
  }
  for (int t = lane; t < DBG_NB * DBG_NB; t += 32) R[bh * 324 + t] = __expf(r[t]);
}

// ---------------------------------------------------------------------------
// Build routed+local buckets: kc/vc[b,h,i, 0..119,:] = sum_j R[i,j]*kb/vb[j],
// kc/vc[b,h,i, 120..239,:] = kb/vb[i]. 288 blocks x 256 threads.
// ---------------------------------------------------------------------------
__global__ __launch_bounds__(256) void route_kernel(
    const bf16* __restrict__ qkv, const float* __restrict__ R,
    bf16* __restrict__ kc, bf16* __restrict__ vc) {
  const int blk = blockIdx.x;                 // (b*8+h)*18+i
  const int i = blk % DBG_NB, bh = blk / DBG_NB;
  const int hh = bh % DBG_H, b = bh / DBG_H;
  __shared__ float rr[DBG_NB];
  if (threadIdx.x < DBG_NB) rr[threadIdx.x] = R[bh * 324 + i * DBG_NB + threadIdx.x];
  __syncthreads();
  bf16* kcb = kc + (size_t)blk * 240 * DBG_DH;
  bf16* vcb = vc + (size_t)blk * 240 * DBG_DH;
  for (int e = threadIdx.x; e < DBG_BS * DBG_DH; e += 256) {
    const int s = e >> 6, d = e & 63;
    float ka = 0.f, va = 0.f;
#pragma unroll
    for (int j = 0; j < DBG_NB; ++j) {
      const size_t idx = ((size_t)(b * DBG_N + j * DBG_BS + s)) * (3 * DBG_D) + hh * DBG_DH + d;
      ka += rr[j] * (float)qkv[idx + DBG_D];
      va += rr[j] * (float)qkv[idx + 2 * DBG_D];
    }
    kcb[e] = (bf16)ka;
    vcb[e] = (bf16)va;
    const size_t sidx = ((size_t)(b * DBG_N + i * DBG_BS + s)) * (3 * DBG_D) + hh * DBG_DH + d;
    kcb[DBG_BS * DBG_DH + e] = qkv[sidx + DBG_D];
    vcb[DBG_BS * DBG_DH + e] = qkv[sidx + 2 * DBG_D];
  }
}

// ---------------------------------------------------------------------------
// Bucket attention. Grid 576 = 288 buckets x 2 halves; block = 128 (4 waves).
// Each wave owns a 16-row q strip: dots = q @ kc^T (WMMA, K=64, 15 N-tiles),
// softmax in registers (shfl within 16-lane halves), attn strip bounced
// through LDS to become a WMMA A operand, then o = attn @ vc (K=240 pad 256).
// ---------------------------------------------------------------------------
__global__ __launch_bounds__(128) void attn_kernel(
    const bf16* __restrict__ qkv, const bf16* __restrict__ kc,
    const bf16* __restrict__ vc, bf16* __restrict__ ao) {
  __shared__ bf16 aLDS[4][16][256];
  const int blk    = blockIdx.x;
  const int half   = blk & 1;
  const int bucket = blk >> 1;                // (b*8+h)*18+i
  const int i = bucket % DBG_NB, bh = bucket / DBG_NB;
  const int hh = bh % DBG_H, b = bh / DBG_H;
  const int wave = threadIdx.x >> 5, lane = threadIdx.x & 31;
  const bf16* kcb = kc + (size_t)bucket * 240 * DBG_DH;
  const bf16* vcb = vc + (size_t)bucket * 240 * DBG_DH;
  const int srow = half * 64 + wave * 16;     // strip base row within bucket

  // ---- Phase 1: dots[16 x 240] = q_strip @ kc^T -------------------------
  v8f d[15] = {};
#pragma unroll
  for (int kt = 0; kt < 2; ++kt) {
    const int k0 = kt * 32;
    v16bf af;
    {
      const int s  = srow + (lane & 15);
      const int kb = k0 + ((lane >> 4) << 3);
      if (s < DBG_BS) {
        const bf16* p = qkv + ((size_t)(b * DBG_N + i * DBG_BS + s)) * (3 * DBG_D) +
                        hh * DBG_DH + kb;
#pragma unroll
        for (int j = 0; j < 8; ++j) {
          const int k = (j < 4) ? (2 * j) : (16 + 2 * (j - 4));
          af[2 * j]     = p[k];
          af[2 * j + 1] = p[k + 1];
        }
      } else {
#pragma unroll
        for (int j = 0; j < 16; ++j) af[j] = (bf16)0.f;
      }
    }
#pragma unroll
    for (int nt = 0; nt < 15; ++nt) {
      const bf16* p = kcb + (size_t)(nt * 16 + (lane & 15)) * DBG_DH + k0 +
                      ((lane >> 4) << 3);
      v16bf bfv;
#pragma unroll
      for (int j = 0; j < 8; ++j) {
        const int k = (j < 4) ? (2 * j) : (16 + 2 * (j - 4));
        bfv[2 * j]     = p[k];
        bfv[2 * j + 1] = p[k + 1];
      }
      d[nt] = __builtin_amdgcn_wmma_f32_16x16x32_bf16(
          false, af, false, bfv, (short)0, d[nt], false, false);
    }
  }

  // ---- Phase 2: row softmax over 240 cols (scale 1/8) -------------------
#pragma unroll
  for (int e = 0; e < 8; ++e) {
    float m = -1e30f;
#pragma unroll
    for (int nt = 0; nt < 15; ++nt) {
      d[nt][e] *= 0.125f;
      m = fmaxf(m, d[nt][e]);
    }
#pragma unroll
    for (int mk = 8; mk >= 1; mk >>= 1) m = fmaxf(m, __shfl_xor(m, mk, 32));
    float s = 0.f;
#pragma unroll
    for (int nt = 0; nt < 15; ++nt) {
      const float ev = __expf(d[nt][e] - m);
      d[nt][e] = ev;
      s += ev;
    }
#pragma unroll
    for (int mk = 8; mk >= 1; mk >>= 1) s += __shfl_xor(s, mk, 32);
    const float inv = 1.f / s;
#pragma unroll
    for (int nt = 0; nt < 15; ++nt) d[nt][e] *= inv;
  }

  // ---- Phase 3: C-layout -> A-layout via LDS ----------------------------
  const int rbase = (lane >> 4) << 3;
#pragma unroll
  for (int nt = 0; nt < 15; ++nt)
#pragma unroll
    for (int e = 0; e < 8; ++e)
      aLDS[wave][rbase + e][nt * 16 + (lane & 15)] = (bf16)d[nt][e];
#pragma unroll
  for (int e = 0; e < 8; ++e)                       // zero-pad K 240..255
    aLDS[wave][rbase + e][240 + (lane & 15)] = (bf16)0.f;
  __syncthreads();

  // ---- Phase 4: o[16 x 64] = attn @ vc (K=240 zero-padded to 256) -------
  v8f oa[4] = {};
#pragma unroll
  for (int kt = 0; kt < 8; ++kt) {
    const v16bf af = frag_rowmajor(&aLDS[wave][0][kt * 32], 256);
#pragma unroll
    for (int nt = 0; nt < 4; ++nt) {
      const int dcol = nt * 16 + (lane & 15);
      const int kb   = kt * 32 + ((lane >> 4) << 3);
      v16bf bfv;
#pragma unroll
      for (int j = 0; j < 8; ++j) {
        const int k = kb + ((j < 4) ? (2 * j) : (16 + 2 * (j - 4)));
        bfv[2 * j]     = (k < 240)     ? vcb[(size_t)k * DBG_DH + dcol]       : (bf16)0.f;
        bfv[2 * j + 1] = (k + 1 < 240) ? vcb[(size_t)(k + 1) * DBG_DH + dcol] : (bf16)0.f;
      }
      oa[nt] = __builtin_amdgcn_wmma_f32_16x16x32_bf16(
          false, af, false, bfv, (short)0, oa[nt], false, false);
    }
  }

  // ---- Phase 5: store o into [B,N,D] bf16 layout for Wo GEMM ------------
#pragma unroll
  for (int nt = 0; nt < 4; ++nt)
#pragma unroll
    for (int e = 0; e < 8; ++e) {
      const int s = srow + rbase + e;
      if (s < DBG_BS)
        ao[((size_t)(b * DBG_N + i * DBG_BS + s)) * DBG_D + hh * DBG_DH +
           nt * 16 + (lane & 15)] = (bf16)oa[nt][e];
    }
}

// ---------------------------------------------------------------------------
// Host-side orchestration
// ---------------------------------------------------------------------------
extern "C" void kernel_launch(void* const* d_in, const int* in_sizes, int n_in,
                              void* d_out, int out_size, void* d_ws, size_t ws_size,
                              hipStream_t stream) {
  (void)in_sizes; (void)n_in; (void)out_size; (void)ws_size;
  const float* x    = (const float*)d_in[0];
  const float* Wp   = (const float*)d_in[1];
  const float* bp   = (const float*)d_in[2];
  const float* pos  = (const float*)d_in[3];
  const float* ln1g = (const float*)d_in[4];
  const float* ln1b = (const float*)d_in[5];
  const float* Wqkv = (const float*)d_in[6];
  const float* Wo   = (const float*)d_in[7];
  const float* bo   = (const float*)d_in[8];
  const float* ln2g = (const float*)d_in[9];
  const float* ln2b = (const float*)d_in[10];
  const float* W1   = (const float*)d_in[11];
  const float* b1   = (const float*)d_in[12];
  const float* W2   = (const float*)d_in[13];
  const float* b2   = (const float*)d_in[14];
  float* h = (float*)d_out;                 // residual stream lives in d_out

  char* ws = (char*)d_ws;
  size_t off = 0;
  auto take = [&](size_t bytes) -> char* {
    char* p = ws + off;
    off += (bytes + 255) & ~(size_t)255;
    return p;
  };
  bf16* wqkv_b = (bf16*)take((size_t)DBG_L * DBG_D * 3 * DBG_D * 2);
  bf16* wo_b   = (bf16*)take((size_t)DBG_L * DBG_D * DBG_D * 2);
  bf16* w1_b   = (bf16*)take((size_t)DBG_L * DBG_D * 4 * DBG_D * 2);
  bf16* w2_b   = (bf16*)take((size_t)DBG_L * 4 * DBG_D * DBG_D * 2);
  bf16* hn     = (bf16*)take((size_t)DBG_M * DBG_D * 2);
  bf16* qkvb   = (bf16*)take((size_t)DBG_M * 3 * DBG_D * 2);
  bf16* aob    = (bf16*)take((size_t)DBG_M * DBG_D * 2);
  bf16* midb   = (bf16*)take((size_t)DBG_M * 4 * DBG_D * 2);
  bf16* kcb    = (bf16*)take((size_t)288 * 240 * DBG_DH * 2);
  bf16* vcb    = (bf16*)take((size_t)288 * 240 * DBG_DH * 2);
  float* sq    = (float*)take((size_t)288 * DBG_DH * 4);
  float* sk    = (float*)take((size_t)288 * DBG_DH * 4);
  float* Rm    = (float*)take((size_t)16 * 324 * 4);

  auto cvt = [&](const float* s, bf16* dst, int n) {
    cvt_kernel<<<(n + 255) / 256, 256, 0, stream>>>(s, dst, n);
  };
  cvt(Wqkv, wqkv_b, DBG_L * DBG_D * 3 * DBG_D);
  cvt(Wo,   wo_b,   DBG_L * DBG_D * DBG_D);
  cvt(W1,   w1_b,   DBG_L * DBG_D * 4 * DBG_D);
  cvt(W2,   w2_b,   DBG_L * 4 * DBG_D * DBG_D);

  patch_kernel<<<DBG_M, 256, 0, stream>>>(x, Wp, bp, pos, h);

  const int mblk = (DBG_M + 63) / 64;       // 68
  for (int l = 0; l < DBG_L; ++l) {
    // --- attention sub-block ---
    ln_kernel<<<DBG_M, 256, 0, stream>>>(h, ln1g + l * DBG_D, ln1b + l * DBG_D, hn);
    gemm_bf16<0><<<dim3(3 * DBG_D / 64, mblk), 128, 0, stream>>>(
        hn, wqkv_b + (size_t)l * DBG_D * 3 * DBG_D, nullptr, qkvb, nullptr,
        DBG_M, 3 * DBG_D, DBG_D);
    summ_kernel<<<288, 64, 0, stream>>>(qkvb, sq, sk);
    sinkhorn_kernel<<<16, 32, 0, stream>>>(sq, sk, Rm);
    route_kernel<<<288, 256, 0, stream>>>(qkvb, Rm, kcb, vcb);
    attn_kernel<<<576, 128, 0, stream>>>(qkvb, kcb, vcb, aob);
    gemm_bf16<2><<<dim3(DBG_D / 64, mblk), 128, 0, stream>>>(
        aob, wo_b + (size_t)l * DBG_D * DBG_D, bo + l * DBG_D, nullptr, h,
        DBG_M, DBG_D, DBG_D);
    // --- MLP sub-block ---
    ln_kernel<<<DBG_M, 256, 0, stream>>>(h, ln2g + l * DBG_D, ln2b + l * DBG_D, hn);
    gemm_bf16<1><<<dim3(4 * DBG_D / 64, mblk), 128, 0, stream>>>(
        hn, w1_b + (size_t)l * DBG_D * 4 * DBG_D, b1 + l * 4 * DBG_D, midb, nullptr,
        DBG_M, 4 * DBG_D, DBG_D);
    gemm_bf16<2><<<dim3(DBG_D / 64, mblk), 128, 0, stream>>>(
        midb, w2_b + (size_t)l * 4 * DBG_D * DBG_D, b2 + l * DBG_D, nullptr, h,
        DBG_M, DBG_D, 4 * DBG_D);
  }
}
